// QSA_30116310679781
// MI455X (gfx1250) — matrix operand order repaired and checked by reading, not compiled, not run on hardware
//
#include <hip/hip_runtime.h>

// ---------------------------------------------------------------------------
// QSA quantum-layer kernel for MI455X (gfx1250, wave32, WMMA).
//
// Heavy math: psi_out = psi @ U.T with real psi => two real GEMMs
// (16384 x 1024 x 1024) vs Re(U), Im(U) done with v_wmma_f32_16x16x32_f16.
// U (4 MB as packed f16 fragments) stays resident in the 192 MB L2; x (64 MB)
// is streamed once from HBM (~2.8 us at 23.3 TB/s) -> GEMM is compute bound.
// The 16 Pauli expectations only touch amplitude bits 0/1, so they fuse into
// the GEMM epilogue via lane shuffles (m^1, m^3 stay inside a 16-wide tile).
// ---------------------------------------------------------------------------

typedef float    v8f  __attribute__((ext_vector_type(8)));
typedef _Float16 v8h  __attribute__((ext_vector_type(8)));
typedef _Float16 v16h __attribute__((ext_vector_type(16)));

#define NQ        10
#define DIM       1024
#define AS        1032            // padded LDS row stride in halves (bank-friendly)
#define BPLANE    (32u*64u*32u*16u)   // halves per plane of Bpack (2 MB)

// ---------------------------------------------------------------------------
// Kernel 1: build U column-by-column and scatter into fragment-ordered Bpack.
// Block c evolves |c> through: 10x (RY(ry0)@RX(rx0)) then 10 layers of
// [10 CNOT(control=j+1 mod 10 bit, target=j bit)] + [10 RY(ry1[j])].
// Bpack[plane][kt][ntile][lane][e] halves; plane0=Re, plane1=Im.
//   k = column c: kt=c>>5, kk=c&31, laneGroup g=kk>>4, element e=kk&15
//   m -> ntile=m>>4, n=m&15, lane=g*16+n   (B 32x16 f16 layout, K split 16/16)
// ---------------------------------------------------------------------------
__global__ __launch_bounds__(256)
void qsa_build_u(const float* __restrict__ rx0, const float* __restrict__ ry0,
                 const float* __restrict__ ry1, _Float16* __restrict__ Bp)
{
    __shared__ float sr[DIM];
    __shared__ float si[DIM];
    const int tid = threadIdx.x;
    const int col = blockIdx.x;

    for (int i = tid; i < DIM; i += 256) { sr[i] = (i == col) ? 1.f : 0.f; si[i] = 0.f; }

    float cy1[NQ], sy1[NQ];
#pragma unroll
    for (int j = 0; j < NQ; ++j) { cy1[j] = __cosf(ry1[j] * 0.5f); sy1[j] = __sinf(ry1[j] * 0.5f); }
    __syncthreads();

    // ----- initial local gates L_j = RY(ry0[j]) @ RX(rx0[j]) on qubit j -----
#pragma unroll
    for (int j = 0; j < NQ; ++j) {
        const float cx = __cosf(rx0[j] * 0.5f), sx = __sinf(rx0[j] * 0.5f);
        const float cy = __cosf(ry0[j] * 0.5f), sy = __sinf(ry0[j] * 0.5f);
        const float L00r =  cy * cx, L00i =  sy * sx;
        const float L01r = -sy * cx, L01i = -cy * sx;
        const float L10r =  sy * cx, L10i = -cy * sx;
        const float L11r =  cy * cx, L11i = -sy * sx;
#pragma unroll
        for (int pp = 0; pp < 2; ++pp) {
            const int p  = tid + pp * 256;                       // pair id 0..511
            const int i0 = ((p >> j) << (j + 1)) | (p & ((1 << j) - 1));
            const int i1 = i0 | (1 << j);
            const float a0r = sr[i0], a0i = si[i0], a1r = sr[i1], a1i = si[i1];
            sr[i0] = L00r*a0r - L00i*a0i + L01r*a1r - L01i*a1i;
            si[i0] = L00r*a0i + L00i*a0r + L01r*a1i + L01i*a1r;
            sr[i1] = L10r*a0r - L10i*a0i + L11r*a1r - L11i*a1i;
            si[i1] = L10r*a0i + L10i*a0r + L11r*a1i + L11i*a1r;
        }
        __syncthreads();
    }

    // ----- 10 entangling layers -----
    for (int l = 0; l < NQ; ++l) {
#pragma unroll
        for (int j = 0; j < NQ; ++j) {                 // CNOTs
            const int cb = (j + 1) % NQ;               // control bit
            const int tb = j;                          // target bit
            const int lo = (cb < tb) ? cb : tb;
            const int hi = (cb < tb) ? tb : cb;
            const int p  = tid;                        // 256 pairs
            const int t1   = ((p >> lo) << (lo + 1)) | (p & ((1 << lo) - 1));
            const int base = ((t1 >> hi) << (hi + 1)) | (t1 & ((1 << hi) - 1));
            const int ia = base | (1 << cb);
            const int ib = ia | (1 << tb);
            const float tr = sr[ia], ti = si[ia];
            sr[ia] = sr[ib]; si[ia] = si[ib];
            sr[ib] = tr;     si[ib] = ti;
            __syncthreads();
        }
#pragma unroll
        for (int j = 0; j < NQ; ++j) {                 // RY(ry1[j])
            const float c = cy1[j], s = sy1[j];
#pragma unroll
            for (int pp = 0; pp < 2; ++pp) {
                const int p  = tid + pp * 256;
                const int i0 = ((p >> j) << (j + 1)) | (p & ((1 << j) - 1));
                const int i1 = i0 | (1 << j);
                const float a0r = sr[i0], a0i = si[i0], a1r = sr[i1], a1i = si[i1];
                sr[i0] = c*a0r - s*a1r;  si[i0] = c*a0i - s*a1i;
                sr[i1] = s*a0r + c*a1r;  si[i1] = s*a0i + c*a1i;
            }
            __syncthreads();
        }
    }

    // ----- scatter column into fragment-ordered f16 Bpack -----
    const int kt = col >> 5, kk = col & 31;
    const int g  = kk >> 4,  e  = kk & 15;
#pragma unroll
    for (int mm = 0; mm < 4; ++mm) {
        const int m  = tid + mm * 256;
        const int nt = m >> 4, n = m & 15;
        const int ln = g * 16 + n;
        const unsigned off = ((((unsigned)kt * 64u + nt) * 32u + ln) * 16u + e);
        Bp[off]          = (_Float16)sr[m];
        Bp[off + BPLANE] = (_Float16)si[m];
    }
}

// ---------------------------------------------------------------------------
// Kernel 2: fused normalize + complex GEMM (WMMA) + Pauli-expectation epilogue.
// One workgroup (8 waves) per 16-row block; each wave owns 8 N-tiles (128 m's).
// ---------------------------------------------------------------------------
__global__ __launch_bounds__(256)
void qsa_gemm(const float* __restrict__ x, const _Float16* __restrict__ Bp,
              float* __restrict__ out)
{
    __shared__ _Float16 Alds[16 * AS];
    __shared__ float    red[256];           // [row 0..15][op 0..15] partial sums

    const int tid  = threadIdx.x;
    const int wg   = blockIdx.x;            // row block: rows wg*16 .. wg*16+15
    const int wave = tid >> 5;
    const int lane = tid & 31;
    const int m16  = lane & 15;
    const int g    = lane >> 4;

    red[tid] = 0.f;

    // -------- phase A: row L2 norms + pack scaled f16 A into LDS --------
    {
        const int r = tid >> 4, seg = tid & 15;
        const float4* xr = (const float4*)(x + ((size_t)(wg * 16 + r)) * DIM + seg * 64);
        float ss = 0.f;
#pragma unroll
        for (int i = 0; i < 16; ++i) {
            float4 v = xr[i];
            ss += v.x*v.x + v.y*v.y + v.z*v.z + v.w*v.w;
        }
#pragma unroll
        for (int d = 1; d < 16; d <<= 1) ss += __shfl_xor(ss, d, 32);
        const float inv = rsqrtf(ss);
#pragma unroll
        for (int i = 0; i < 16; ++i) {
            float4 v = xr[i];                       // L2/L0 hit (second pass)
            _Float16* a = &Alds[r * AS + seg * 64 + i * 4];
            a[0] = (_Float16)(v.x * inv);
            a[1] = (_Float16)(v.y * inv);
            a[2] = (_Float16)(v.z * inv);
            a[3] = (_Float16)(v.w * inv);
        }
    }
    __syncthreads();

    // -------- phase B: K-loop, 16 WMMAs (8 tiles x Re/Im) per step --------
    const v8f vz = {0.f,0.f,0.f,0.f,0.f,0.f,0.f,0.f};
    v8f accR[8], accI[8];
#pragma unroll
    for (int t = 0; t < 8; ++t) { accR[t] = vz; accI[t] = vz; }

    for (int kb = 0; kb < DIM; kb += 32) {
        const int kt = kb >> 5;
        // A fragment: contiguous K split matching Bpack (K = kb+16g+0..15)
        const v8h alo = *(const v8h*)&Alds[m16 * AS + kb + 16 * g];
        const v8h ahi = *(const v8h*)&Alds[m16 * AS + kb + 16 * g + 8];
        const v16h af = __builtin_shufflevector(alo, ahi,
                          0,1,2,3,4,5,6,7,8,9,10,11,12,13,14,15);
#pragma unroll
        for (int t = 0; t < 8; ++t) {
            const int nt = wave * 8 + t;
            const unsigned off = (((unsigned)kt * 64u + nt) * 32u + lane) * 16u;
            const v16h br = *(const v16h*)(Bp + off);            // Re(U) frag (L2)
            const v16h bi = *(const v16h*)(Bp + off + BPLANE);   // Im(U) frag (L2)
            accR[t] = __builtin_amdgcn_wmma_f32_16x16x32_f16(
                          false, af, false, br, (short)0, accR[t], false, false);
            accI[t] = __builtin_amdgcn_wmma_f32_16x16x32_f16(
                          false, af, false, bi, (short)0, accI[t], false, false);
        }
    }

    // -------- phase C: Pauli expectation epilogue --------
    // D layout: VGPR j, lane L -> row = j + 8*(L>>4), m = tile*16 + (L&15).
    // All pairings flip bits 0/1 of m -> shfl_xor 1 / 3 within the 16-lane group.
#pragma unroll
    for (int j = 0; j < 8; ++j) {
        float pop[16];
#pragma unroll
        for (int o = 0; o < 16; ++o) pop[o] = 0.f;
#pragma unroll
        for (int t = 0; t < 8; ++t) {
            const int m  = (wave * 8 + t) * 16 + m16;
            const float ar = accR[t][j], ai = accI[t][j];
            const float br = __shfl_xor(ar, 3, 32), bi = __shfl_xor(ai, 3, 32); // phi[m^3]
            const float cr = __shfl_xor(ar, 1, 32), ci = __shfl_xor(ai, 1, 32); // phi[m^1]
            const float w  = ar*ar + ai*ai;
            const int b0 = m & 1, b1 = (m >> 1) & 1;
#pragma unroll
            for (int q = 0; q < NQ; ++q)                   // Z_q strings
                pop[q] += ((m >> q) & 1) ? -w : w;
            const float dxx = ar*br + ai*bi;               // Re(conj a * b)
            pop[10] += dxx;                                // XX
            pop[11] += (b0 == b1) ? -dxx : dxx;            // YY
            pop[12] += (b0 ^ b1) ? -w : w;                 // ZZ
            const float dxy = ai*br - ar*bi;               // Re(conj a * i b)
            pop[13] += b1 ? dxy : -dxy;                    // XY
            const float dxz = ar*cr + ai*ci;
            pop[14] += b1 ? -dxz : dxz;                    // XZ
            const float dyz = ai*cr - ar*ci;
            pop[15] += (b0 ^ b1) ? dyz : -dyz;             // YZ
        }
#pragma unroll
        for (int o = 0; o < 16; ++o) {
#pragma unroll
            for (int d = 1; d < 16; d <<= 1) pop[o] += __shfl_xor(pop[o], d, 32);
        }
        if (m16 == 0) {                                    // lanes 0 and 16
            const int row = j + 8 * g;
#pragma unroll
            for (int o = 0; o < 16; ++o) atomicAdd(&red[row * 16 + o], pop[o]);
        }
    }
    __syncthreads();

    // out[(wg*16 + row)*16 + op] == out[wg*256 + tid]
    out[(size_t)wg * 256 + tid] = red[tid];
}

// ---------------------------------------------------------------------------
extern "C" void kernel_launch(void* const* d_in, const int* in_sizes, int n_in,
                              void* d_out, int out_size, void* d_ws, size_t ws_size,
                              hipStream_t stream)
{
    const float* x   = (const float*)d_in[0];   // (16,1024,1024) f32
    const float* rx0 = (const float*)d_in[1];   // (10,)
    const float* ry0 = (const float*)d_in[2];   // (10,)
    const float* ry1 = (const float*)d_in[3];   // (10,)
    float* out = (float*)d_out;                 // (16,1024,16) f32

    _Float16* Bpack = (_Float16*)d_ws;          // needs 4 MB of workspace

    qsa_build_u<<<dim3(DIM), dim3(256), 0, stream>>>(rx0, ry0, ry1, Bpack);
    qsa_gemm  <<<dim3(16384 / 16), dim3(256), 0, stream>>>(x, Bpack, out);
}
